// NCA2D_46042049413431
// MI455X (gfx1250) — matrix operand order repaired and checked by reading, not compiled, not run on hardware
//
#include <hip/hip_runtime.h>
#include <math.h>

// Problem constants (match reference)
#define NSTEPS 64
#define BB 16
#define HH 128
#define WW 128
#define CC 16
#define TT 8
#define HID 128

#define TILE_H 8
#define TILE_W 16
#define NTHREADS 256   // 8 wave32s; wave w owns tile row w (16 pixels = one M=16 WMMA tile)

typedef __attribute__((ext_vector_type(16))) _Float16 v16h;
typedef __attribute__((ext_vector_type(8)))  float    v8f;
typedef __attribute__((ext_vector_type(4)))  unsigned int u32x4;
typedef __attribute__((ext_vector_type(4)))  int      i32x4;
typedef __attribute__((ext_vector_type(8)))  int      i32x8;

// Padded f16 weight images (built once per launch by nca_prep_kernel)
#define W0K 68      // FC0 K padded 49 -> 64 (+4 stride pad)
#define W1K 136     // FC1 K padded 128 (+8 stride pad)
#define FP_B0 0     // fparams float layout
#define FP_CW 128   // 2 x 9 x 16 conv taps
#define FP_CB 416   // 2 x 16 conv biases
#define FP_N  448

__global__ void nca_zero_kernel(float* __restrict__ p, long n) {
    long i = (long)blockIdx.x * blockDim.x + threadIdx.x;
    long stride = (long)gridDim.x * blockDim.x;
    for (; i < n; i += stride) p[i] = 0.0f;
}

// One-time weight conversion: f32 -> f16, K zero-padded, conv taps packed.
__global__ void nca_prep_kernel(const float* __restrict__ fc0w,
                                const float* __restrict__ fc1w,
                                const float* __restrict__ fc0b,
                                const float* __restrict__ p0w,
                                const float* __restrict__ p0b,
                                const float* __restrict__ p1w,
                                const float* __restrict__ p1b,
                                _Float16* __restrict__ w0h,
                                _Float16* __restrict__ w1h,
                                float* __restrict__ fparams) {
    int i = blockIdx.x * blockDim.x + threadIdx.x;
    int stride = gridDim.x * blockDim.x;
    for (int j = i; j < 128 * W0K; j += stride) {
        int k = j % W0K, n = j / W0K;
        w0h[j] = (k < 49) ? (_Float16)fc0w[n * 49 + k] : (_Float16)0.0f;
    }
    for (int j = i; j < 16 * W1K; j += stride) {
        int k = j % W1K, n = j / W1K;
        w1h[j] = (k < 128) ? (_Float16)fc1w[n * 128 + k] : (_Float16)0.0f;
    }
    for (int j = i; j < HID; j += stride) fparams[FP_B0 + j] = fc0b[j];
    for (int j = i; j < 144; j += stride) {
        fparams[FP_CW + j]       = p0w[j];   // (dy*3+dx)*16 + c
        fparams[FP_CW + 144 + j] = p1w[j];
    }
    for (int j = i; j < 16; j += stride) {
        fparams[FP_CB + j]      = p0b[j];
        fparams[FP_CB + 16 + j] = p1b[j];
    }
}

__global__ __launch_bounds__(NTHREADS)
void nca_step_kernel(const float* __restrict__ x_src,
                     float* __restrict__ x_dst,
                     float* __restrict__ out,              // [B,T,H,W,C]
                     const int* __restrict__ acq,          // [B,T]
                     const _Float16* __restrict__ w0h,     // [128][W0K]
                     const _Float16* __restrict__ w1h,     // [16][W1K]
                     const float* __restrict__ fparams,    // b0 | conv taps | conv biases
                     int step)
{
    // LDS (~86 KB): strides padded off powers of two for conflict-free fragment gathers
    __shared__ float    s_tile[TILE_H + 2][TILE_W + 2][CC]; // contiguous 10 x 288 f32
    __shared__ _Float16 s_feat[128][72];
    __shared__ _Float16 s_h[128][136];
    __shared__ _Float16 s_w0[128][W0K];
    __shared__ _Float16 s_w1[16][W1K];
    __shared__ float    s_fp[FP_N];

    const int tid  = threadIdx.x;
    const int wv   = tid >> 5;
    const int lane = tid & 31;
    const int col  = lane & 15;
    const int g    = lane >> 4;
    const int b    = blockIdx.z;
    const int gy0  = blockIdx.y * TILE_H;
    const int gx0  = blockIdx.x * TILE_W;
    const int prow = wv * 16;

    // ---- stage x tile: TDM for interior blocks, manual reflect path at borders ----
    const bool interior = (gy0 >= 1) && (gy0 + TILE_H + 1 <= HH) &&
                          (gx0 >= 1) && (gx0 + TILE_W + 1 <= WW);
    if (interior) {
        if (wv == 0) {
            // 2D tile: 10 lines of 288 f32, global line stride 2048 elements,
            // contiguous into s_tile. D# per CDNA5 ISA 8.3/8.4.
            unsigned lds_addr = (unsigned)(unsigned long long)(uintptr_t)&s_tile[0][0][0];
            const float* gsrc = x_src + ((size_t)(b * HH + gy0 - 1) * WW + (gx0 - 1)) * CC;
            unsigned long long ga = (unsigned long long)(uintptr_t)gsrc;
            u32x4 d0;
            d0[0] = 1u;                                          // count=1, user mode
            d0[1] = lds_addr;                                    // lds_addr
            d0[2] = (unsigned)ga;                                // global_addr[31:0]
            d0[3] = (unsigned)((ga >> 32) & 0x01FFFFFFull) | 0x80000000u; // addr[56:32] | type=2
            i32x8 d1;
            d1[0] = 0x00020000;                 // data_size = 4B
            d1[1] = (int)(2048u << 16);         // tensor_dim0 = W*C = 2048 (lo16 in [63:48])
            d1[2] = (int)(128u  << 16);         // tensor_dim0 hi=0 | tensor_dim1 = 128 lo16
            d1[3] = (int)(288u  << 16);         // tensor_dim1 hi=0 | tile_dim0 = 288
            d1[4] = 10;                         // tile_dim1 = 10, tile_dim2 = 0
            d1[5] = 2048;                       // tensor_dim0_stride lo32
            d1[6] = 0;
            d1[7] = 0;
            i32x4 dz; dz[0] = 0; dz[1] = 0; dz[2] = 0; dz[3] = 0;
#if __clang_major__ >= 23
            i32x8 dz8; dz8[0]=0; dz8[1]=0; dz8[2]=0; dz8[3]=0; dz8[4]=0; dz8[5]=0; dz8[6]=0; dz8[7]=0;
            __builtin_amdgcn_tensor_load_to_lds(d0, d1, dz, dz, dz8, 0);
#else
            __builtin_amdgcn_tensor_load_to_lds(d0, d1, dz, dz, 0);
#endif
        }
    } else {
        for (int i = tid; i < (TILE_H + 2) * (TILE_W + 2) * CC; i += NTHREADS) {
            int c  = i & (CC - 1);
            int xy = i / CC;
            int tx = xy % (TILE_W + 2);
            int ty = xy / (TILE_W + 2);
            int gy = gy0 + ty - 1;
            int gx = gx0 + tx - 1;
            gy = gy < 0 ? -gy : (gy >= HH ? 2 * HH - 2 - gy : gy);
            gx = gx < 0 ? -gx : (gx >= WW ? 2 * WW - 2 - gx : gx);
            s_tile[ty][tx][c] = x_src[((b * HH + gy) * WW + gx) * CC + c];
        }
    }

    // ---- stage preconverted weights: straight dword copies (overlaps the DMA) ----
    {
        const unsigned* src0 = (const unsigned*)w0h;
        unsigned* dst0 = (unsigned*)&s_w0[0][0];
        for (int i = tid; i < 128 * W0K / 2; i += NTHREADS) dst0[i] = src0[i];
        const unsigned* src1 = (const unsigned*)w1h;
        unsigned* dst1 = (unsigned*)&s_w1[0][0];
        for (int i = tid; i < 16 * W1K / 2; i += NTHREADS) dst1[i] = src1[i];
        for (int i = tid; i < FP_N; i += NTHREADS) s_fp[i] = fparams[i];
    }
    if (interior && wv == 0) __builtin_amdgcn_s_wait_tensorcnt(0);
    __syncthreads();

    // ---- perceive: identity + 2 depthwise 3x3 -> f16 feature rows ----
    const float prog = (float)step * (1.0f / (float)NSTEPS);
    {
        const int ch0 = g * 8;
        #pragma unroll
        for (int cc = 0; cc < 8; ++cc) {
            int c = ch0 + cc;
            float xv = s_tile[wv + 1][col + 1][c];
            float z0 = s_fp[FP_CB + c], z1 = s_fp[FP_CB + 16 + c];
            #pragma unroll
            for (int dy = 0; dy < 3; ++dy)
                #pragma unroll
                for (int dx = 0; dx < 3; ++dx) {
                    float xn = s_tile[wv + dy][col + dx][c];
                    z0 = fmaf(xn, s_fp[FP_CW + (dy * 3 + dx) * 16 + c], z0);
                    z1 = fmaf(xn, s_fp[FP_CW + 144 + (dy * 3 + dx) * 16 + c], z1);
                }
            s_feat[prow + col][c]      = (_Float16)xv;
            s_feat[prow + col][16 + c] = (_Float16)z0;
            s_feat[prow + col][32 + c] = (_Float16)z1;
        }
        if (g == 0) {
            s_feat[prow + col][48] = (_Float16)prog;   // time-progress channel
            #pragma unroll
            for (int k = 49; k < 56; ++k) s_feat[prow + col][k] = (_Float16)0.0f;
        } else {
            #pragma unroll
            for (int k = 56; k < 64; ++k) s_feat[prow + col][k] = (_Float16)0.0f;
        }
    }
    __syncthreads();

    // ---- FC0: [16 px x 64] x [64 x 128] via 8 N-tiles * 2 K-chunks of WMMA ----
    v16h a0, a1;
    #pragma unroll
    for (int j = 0; j < 16; ++j) {
        int k = 16 * g + j;                     // A 16x32: half j -> K = 16g + j
        a0[j] = s_feat[prow + col][k];
        a1[j] = s_feat[prow + col][k + 32];
    }

    #pragma unroll
    for (int nt = 0; nt < 8; ++nt) {
        v16h bf0, bf1;                          // B 32x16: half i -> K = 16g + i
        #pragma unroll
        for (int i = 0; i < 16; ++i) {
            bf0[i] = s_w0[nt * 16 + col][g * 16 + i];
            bf1[i] = s_w0[nt * 16 + col][32 + g * 16 + i];
        }
        float bias = s_fp[FP_B0 + nt * 16 + col];
        v8f acc;
        #pragma unroll
        for (int r = 0; r < 8; ++r) acc[r] = bias;
        acc = __builtin_amdgcn_wmma_f32_16x16x32_f16(false, a0, false, bf0,
                                                     (short)0, acc, false, false);
        acc = __builtin_amdgcn_wmma_f32_16x16x32_f16(false, a1, false, bf1,
                                                     (short)0, acc, false, false);
        #pragma unroll
        for (int r = 0; r < 8; ++r) {           // ReLU + relayout D into A-major h
            float hv = acc[r] > 0.0f ? acc[r] : 0.0f;
            s_h[prow + r + 8 * g][nt * 16 + col] = (_Float16)hv;
        }
    }
    __syncthreads();

    // ---- FC1: [16 px x 128] x [128 x 16] via 4 K-chunks of WMMA ----
    v8f acc2 = {};
    #pragma unroll
    for (int kc = 0; kc < 4; ++kc) {
        v16h ah, bh;
        #pragma unroll
        for (int j = 0; j < 16; ++j) {
            int k = 16 * g + j;
            ah[j] = s_h[prow + col][kc * 32 + k];
        }
        #pragma unroll
        for (int i = 0; i < 16; ++i)
            bh[i] = s_w1[col][kc * 32 + g * 16 + i];
        acc2 = __builtin_amdgcn_wmma_f32_16x16x32_f16(false, ah, false, bh,
                                                      (short)0, acc2, false, false);
    }

    // ---- sigmoid + residual update + snapshot scatter ----
    unsigned smask = 0;
    if (step > 0) {
        #pragma unroll
        for (int t = 0; t < TT; ++t)
            if (acq[b * TT + t] == step) smask |= (1u << t);
    }

    const int gy = gy0 + wv;
    #pragma unroll
    for (int r = 0; r < 8; ++r) {
        int px = r + 8 * g;                     // pixel column (D-row M)
        int gx = gx0 + px;
        float xold = s_tile[wv + 1][px + 1][col];
        float dx = 1.0f / (1.0f + __expf(-acc2[r]));
        float xnew = xold + dx;
        x_dst[((b * HH + gy) * WW + gx) * CC + col] = xnew;   // reread next step: keep RT
        for (int t = 0; t < TT; ++t)
            if (smask & (1u << t))                             // write-once: non-temporal
                __builtin_nontemporal_store(
                    xnew, &out[(((b * TT + t) * HH + gy) * WW + gx) * CC + col]);
    }
}

extern "C" void kernel_launch(void* const* d_in, const int* in_sizes, int n_in,
                              void* d_out, int out_size, void* d_ws, size_t ws_size,
                              hipStream_t stream) {
    const float* x0   = (const float*)d_in[0];
    const int*   acq  = (const int*)d_in[1];
    const float* p0w  = (const float*)d_in[2];
    const float* p0b  = (const float*)d_in[3];
    const float* p1w  = (const float*)d_in[4];
    const float* p1b  = (const float*)d_in[5];
    const float* fc0w = (const float*)d_in[6];
    const float* fc0b = (const float*)d_in[7];
    const float* fc1w = (const float*)d_in[8];
    float* out = (float*)d_out;

    const size_t state = (size_t)BB * HH * WW * CC;   // 4M floats = 16MB
    char* ws = (char*)d_ws;
    float* buf0 = (float*)ws;
    float* buf1 = buf0 + state;
    _Float16* w0h = (_Float16*)(ws + 2 * state * sizeof(float));
    _Float16* w1h = w0h + 128 * W0K;
    float* fparams = (float*)(w1h + 16 * W1K);

    // data0 = zeros (d_out is poisoned by the harness)
    nca_zero_kernel<<<1024, 256, 0, stream>>>(out, (long)out_size);
    // one-time weight conversion into L2-resident padded f16 images
    nca_prep_kernel<<<64, 256, 0, stream>>>(fc0w, fc1w, fc0b, p0w, p0b, p1w, p1b,
                                            w0h, w1h, fparams);

    dim3 grid(WW / TILE_W, HH / TILE_H, BB);          // 8 x 16 x 16 = 2048 WGs/step
    for (int s = 0; s < NSTEPS; ++s) {
        const float* src = (s == 0) ? x0 : ((s & 1) ? buf0 : buf1);
        float* dst = (s & 1) ? buf1 : buf0;
        nca_step_kernel<<<grid, NTHREADS, 0, stream>>>(
            src, dst, out, acq, w0h, w1h, fparams, s);
    }
}
